// K_attention_12086037971047
// MI455X (gfx1250) — matrix operand order repaired and checked by reading, not compiled, not run on hardware
//
#include <hip/hip_runtime.h>

// Gaussian-kernel self-attention:  out = x + (exp(-d2*sigma) ⊙ offdiag) @ x
// B=16, T=2048, C=64, fp32 in/out, f16 WMMA compute with f32 accumulate.

typedef __attribute__((ext_vector_type(16))) _Float16 v16h;
typedef __attribute__((ext_vector_type(8)))  _Float16 v8h;
typedef __attribute__((ext_vector_type(8)))  float    v8f;

#define WMMA_F16F32(A, B, C) \
  __builtin_amdgcn_wmma_f32_16x16x32_f16(false, (A), false, (B), (short)0, (C), false, false)

static constexpr int Bn = 16;
static constexpr int Tn = 2048;
static constexpr int Cn = 64;

__global__ __launch_bounds__(128, 1)
void gauss_attn_kernel(const float* __restrict__ x,
                       const float* __restrict__ r_sigma,
                       float* __restrict__ out)
{
    // LDS staging:  j-block of 32 rows, both layouts, shared by 4 waves.
    __shared__ alignas(32) _Float16 ldsN[32][64];     // [j_local][c]   4 KB
    __shared__ alignas(32) _Float16 ldsT[64][32];     // [c][j_local]   4 KB
    __shared__ alignas(16) float    sqj[32];          // per-j ||x_j||^2
    __shared__ alignas(32) _Float16 sbuf[4][16][32];  // per-wave S transpose 4 KB

    const int tid  = threadIdx.x;
    const int lane = tid & 31;
    const int wave = tid >> 5;
    const int m    = lane & 15;   // column index in B/C/D fragments, row in A frag
    const int half = lane >> 4;   // K-half selector per ISA layouts

    const int bIdx = blockIdx.x >> 5;                 // 32 blocks per batch
    const int i0   = (blockIdx.x & 31) * 64 + wave * 16;
    const int jdiag = i0 & ~31;                       // the single j-block holding diag

    // Fold sigma and log2(e):  km = exp2( min( (sqi+sqj)*nc - 2*nc*gram, 0 ) )
    const float rs   = r_sigma[0];
    const float nc   = rs * -1.44269504088896340736f; // -sigma * log2(e)  (< 0)
    const float m2nc = -2.0f * nc;

    const float* xb = x + (size_t)bIdx * Tn * Cn;

    // ---------------- A fragments: rows i0..i0+15 of X, f32 -> f16 ----------
    // A layout (16x32 f16): lane l holds row m=l&15; v0..7 = K [16*half, +16).
    // a0 covers c in [0,32), a1 covers [32,64).
    const float* rowp = xb + (size_t)(i0 + m) * Cn;
    const int abase = 16 * half;
    v16h a0, a1;
    float sqpart = 0.0f;
    {
        const float4* r4 = (const float4*)(rowp);
        float4 f0a = r4[abase / 4 + 0], f0b = r4[abase / 4 + 1];   // c = abase..+8
        float4 f1a = r4[2 + abase / 4], f1b = r4[3 + abase / 4];   // c = 8+abase..
        float4 f2a = r4[8 + abase / 4], f2b = r4[9 + abase / 4];   // c = 32+abase..
        float4 f3a = r4[10 + abase / 4], f3b = r4[11 + abase / 4]; // c = 40+abase..
        float c0[8] = {f0a.x, f0a.y, f0a.z, f0a.w, f0b.x, f0b.y, f0b.z, f0b.w};
        float c1[8] = {f1a.x, f1a.y, f1a.z, f1a.w, f1b.x, f1b.y, f1b.z, f1b.w};
        float c2[8] = {f2a.x, f2a.y, f2a.z, f2a.w, f2b.x, f2b.y, f2b.z, f2b.w};
        float c3[8] = {f3a.x, f3a.y, f3a.z, f3a.w, f3b.x, f3b.y, f3b.z, f3b.w};
#pragma unroll
        for (int k = 0; k < 8; ++k) {
            _Float16 h0 = (_Float16)c0[k], h1 = (_Float16)c1[k];
            _Float16 h2 = (_Float16)c2[k], h3 = (_Float16)c3[k];
            a0[k] = h0; a0[8 + k] = h1; a1[k] = h2; a1[8 + k] = h3;
            sqpart += (float)h0 * (float)h0 + (float)h1 * (float)h1
                    + (float)h2 * (float)h2 + (float)h3 * (float)h3;
        }
    }
    // Full ||x_i||^2 for row m: combine the two K-half lanes, broadcast to the
    // C/D layout (vgpr r holds M = r + 8*half), pre-scale by nc.
    float sqrow = sqpart + __shfl_xor(sqpart, 16);
    float sqinc[8];
#pragma unroll
    for (int r = 0; r < 8; ++r) sqinc[r] = __shfl(sqrow, r + 8 * half) * nc;

    v8f acc[4];
#pragma unroll
    for (int t = 0; t < 4; ++t) acc[t] = (v8f){0, 0, 0, 0, 0, 0, 0, 0};

    // staging role of this thread: row j_local = tid/4, 16-wide c segment
    const int jrow = tid >> 2;
    const int cseg = (tid & 3) * 16;

    for (int j0 = 0; j0 < Tn; j0 += 32) {
        __syncthreads();   // previous iteration's LDS reads complete

        // ---------------- stage 32 j-rows into LDS (both layouts) ----------
        {
            const float* jp = xb + (size_t)(j0 + jrow) * Cn + cseg;
            const float4* jp4 = (const float4*)jp;
            float4 f0 = jp4[0], f1 = jp4[1], f2 = jp4[2], f3 = jp4[3];
            float fv[16] = {f0.x, f0.y, f0.z, f0.w, f1.x, f1.y, f1.z, f1.w,
                            f2.x, f2.y, f2.z, f2.w, f3.x, f3.y, f3.z, f3.w};
            v8h h0, h1;
            float sp = 0.0f;
#pragma unroll
            for (int k = 0; k < 8; ++k) {
                _Float16 ha = (_Float16)fv[k], hb = (_Float16)fv[8 + k];
                h0[k] = ha; h1[k] = hb;
                sp += (float)ha * (float)ha + (float)hb * (float)hb;
            }
            *(v8h*)&ldsN[jrow][cseg]     = h0;
            *(v8h*)&ldsN[jrow][cseg + 8] = h1;
#pragma unroll
            for (int k = 0; k < 8; ++k) {
                ldsT[cseg + k][jrow]     = h0[k];
                ldsT[cseg + 8 + k][jrow] = h1[k];
            }
            sp += __shfl_xor(sp, 1);
            sp += __shfl_xor(sp, 2);
            if ((tid & 3) == 0) sqj[jrow] = sp;
            if (j0 + 32 < Tn) __builtin_prefetch(jp + 32 * Cn, 0, 1);
        }
        __syncthreads();

        // ---------------- gram: 2 j-tiles, interleaved accumulators --------
        // Load ALL four B fragments into distinct registers first, then issue
        // the 4 WMMAs with alternating (independent) accumulators so the
        // matrix pipe is not gated on per-load dscnt drains.
        v16h bg0, bg1, bg2, bg3;
        bg0 = *(const v16h*)&ldsN[m][16 * half];            // tile0, c [0,32)
        bg1 = *(const v16h*)&ldsN[m][32 + 16 * half];       // tile0, c [32,64)
        bg2 = *(const v16h*)&ldsN[16 + m][16 * half];       // tile1, c [0,32)
        bg3 = *(const v16h*)&ldsN[16 + m][32 + 16 * half];  // tile1, c [32,64)

        v8f g0 = (v8f){0, 0, 0, 0, 0, 0, 0, 0};
        v8f g1 = (v8f){0, 0, 0, 0, 0, 0, 0, 0};
        g0 = WMMA_F16F32(a0, bg0, g0);
        g1 = WMMA_F16F32(a0, bg2, g1);
        g0 = WMMA_F16F32(a1, bg1, g0);
        g1 = WMMA_F16F32(a1, bg3, g1);

        // ---------------- exp epilogue -> S (f16 in sbuf) ------------------
        const float sj0 = sqj[m]      * nc;
        const float sj1 = sqj[16 + m] * nc;
        float km0[8], km1[8];
#pragma unroll
        for (int r = 0; r < 8; ++r) {
            // arg = nc*d2 ; clamp d2>=0  <=>  arg<=0 (nc<0)
            float a = fminf(fmaf(m2nc, g0[r], sqinc[r] + sj0), 0.0f);
            float b = fminf(fmaf(m2nc, g1[r], sqinc[r] + sj1), 0.0f);
            km0[r] = __builtin_amdgcn_exp2f(a);
            km1[r] = __builtin_amdgcn_exp2f(b);
        }
        // Diagonal only when this j-block overlaps the i-tile: wave-uniform
        // branch, taken in exactly 1 of 64 iterations.
        if (j0 == jdiag) {
#pragma unroll
            for (int r = 0; r < 8; ++r) {
                if (i0 + r + 8 * half == j0 + m)      km0[r] = 0.0f;
                if (i0 + r + 8 * half == j0 + 16 + m) km1[r] = 0.0f;
            }
        }
#pragma unroll
        for (int r = 0; r < 8; ++r) {
            sbuf[wave][r + 8 * half][m]      = (_Float16)km0[r];
            sbuf[wave][r + 8 * half][16 + m] = (_Float16)km1[r];
        }
        // wave-internal LDS transpose: stores above feed loads below
        asm volatile("s_wait_dscnt 0" ::: "memory");

        // A fragment of S (16x32 f16), same lane layout as a0/a1
        v16h sA;
        {
            v8h lo = *(const v8h*)&sbuf[wave][m][16 * half];
            v8h hi = *(const v8h*)&sbuf[wave][m][16 * half + 8];
#pragma unroll
            for (int k = 0; k < 8; ++k) { sA[k] = lo[k]; sA[8 + k] = hi[k]; }
        }

        // ---------------- out += S @ X_j  (K = 32 j's, 4 c-tiles) ----------
        // Distinct fragment registers -> one dscnt wait, 4 back-to-back WMMAs.
        v16h b2_0 = *(const v16h*)&ldsT[m][16 * half];
        v16h b2_1 = *(const v16h*)&ldsT[16 + m][16 * half];
        v16h b2_2 = *(const v16h*)&ldsT[32 + m][16 * half];
        v16h b2_3 = *(const v16h*)&ldsT[48 + m][16 * half];
        acc[0] = WMMA_F16F32(sA, b2_0, acc[0]);
        acc[1] = WMMA_F16F32(sA, b2_1, acc[1]);
        acc[2] = WMMA_F16F32(sA, b2_2, acc[2]);
        acc[3] = WMMA_F16F32(sA, b2_3, acc[3]);
    }

    // ---------------- epilogue: out = x + acc -------------------------------
#pragma unroll
    for (int tc = 0; tc < 4; ++tc) {
#pragma unroll
        for (int r = 0; r < 8; ++r) {
            const int mrow = r + 8 * half;
            const size_t idx =
                ((size_t)bIdx * Tn + (i0 + mrow)) * Cn + tc * 16 + m;
            out[idx] = x[idx] + acc[tc][r];
        }
    }
}

extern "C" void kernel_launch(void* const* d_in, const int* in_sizes, int n_in,
                              void* d_out, int out_size, void* d_ws, size_t ws_size,
                              hipStream_t stream)
{
    (void)in_sizes; (void)n_in; (void)out_size; (void)d_ws; (void)ws_size;
    const float* x  = (const float*)d_in[0];
    const float* rs = (const float*)d_in[1];
    float* out      = (float*)d_out;

    // 16 batches * 32 blocks; each block = 4 waves, each wave = one 16-row tile
    dim3 grid(Bn * (Tn / 64));
    dim3 block(128);
    hipLaunchKernelGGL(gauss_attn_kernel, grid, block, 0, stream, x, rs, out);
}
